// TrafSparseGATLayer_19971597926810
// MI455X (gfx1250) — matrix operand order repaired and checked by reading, not compiled, not run on hardware
//
#include <hip/hip_runtime.h>
#include <math.h>

// Problem constants (from reference)
#define BDIM   8
#define TDIM   12
#define BT     (BDIM*TDIM)        // 96
#define NNODE  1000
#define FIN    64
#define HH     8
#define FOUT   16
#define COLS   (HH*FOUT)          // 128
#define EDGES  8000
#define ROWS   (BT*NNODE)         // 96000
#define NH     (NNODE*HH)         // 8000 per bt
#define TOTAL_OUT (ROWS*COLS)     // 12,288,000
#define TOTAL_EH  (BT*EDGES*HH)   // 6,144,000
#define TOTAL_NHB (BT*NNODE*HH)   // 768,000

typedef __attribute__((ext_vector_type(16))) _Float16 v16h;
typedef __attribute__((ext_vector_type(8)))  float    v8f;

// ---------------------------------------------------------------------------
// init: zero softmax denominators and the global-max cell
// ---------------------------------------------------------------------------
__global__ void k_init(float* __restrict__ denom, unsigned* __restrict__ gmax) {
    int i = blockIdx.x * blockDim.x + threadIdx.x;
    if (i < TOTAL_NHB) denom[i] = 0.0f;
    if (i == 0) *gmax = 0u;   // ordered-uint encoding identity for max
}

// ---------------------------------------------------------------------------
// gemm: P = X * Wp^T  (cols 0..127 -> proj),  S = X * Ws^T (cols 128..255 -> out)
// X: [ROWS, 64] f32, Wp/Ws: [128, 64] f32 row-major.
// Block = 128 threads (4 waves). grid = (ROWS/16, 4).
// Wave w handles col-tile (blockIdx.y*4 + w), 16x16 f32 output tile,
// K=64 split into two K=32 v_wmma_f32_16x16x32_f16 steps.
// ---------------------------------------------------------------------------
__global__ void k_gemm(const float* __restrict__ X,
                       const float* __restrict__ Wp,
                       const float* __restrict__ Ws,
                       float* __restrict__ proj,
                       float* __restrict__ outAcc) {
    const int wave = threadIdx.x >> 5;
    const int lane = threadIdx.x & 31;
    const int rowBase  = blockIdx.x * 16;
    const int colTile  = blockIdx.y * 4 + wave;   // 0..15
    const int colBase  = colTile * 16;            // 0..240

    // A-operand lane mapping (16-bit A 16x32): row = lane%16,
    //   elements 0..7  hold K = kbase..kbase+7        (kbase = (lane/16)*8)
    //   elements 8..15 hold K = 16+kbase..16+kbase+7
    const int arow  = rowBase + (lane & 15);
    const int kbase = (lane >> 4) * 8;
    const float* xrow = X + (size_t)arow * FIN;

    // B-operand lane mapping (16-bit B 32x16): col = lane%16,
    //   elements 0..7 hold K = koff/2..koff/2+7, elements 8..15 hold K = 16+koff/2..16+koff/2+7 (koff = (lane/16)*16)
    const int ncol  = colBase + (lane & 15);
    const int koff  = (lane >> 4) * 16;
    const float* wrow = (colBase < COLS) ? (Wp + (size_t)ncol * FIN)
                                         : (Ws + (size_t)(ncol - COLS) * FIN);

    v8f acc = {};
    #pragma unroll
    for (int k0 = 0; k0 < FIN; k0 += 32) {
        v16h a, b;
        #pragma unroll
        for (int j = 0; j < 8; ++j) {
            a[j]     = (_Float16)xrow[k0 + kbase + j];
            a[j + 8] = (_Float16)xrow[k0 + 16 + kbase + j];
        }
        #pragma unroll
        for (int j = 0; j < 8; ++j) {
            b[j]     = (_Float16)wrow[k0 + (koff >> 1) + j];
            b[j + 8] = (_Float16)wrow[k0 + 16 + (koff >> 1) + j];
        }
        acc = __builtin_amdgcn_wmma_f32_16x16x32_f16(
                  /*neg_a=*/false, a, /*neg_b=*/false, b,
                  /*c_mod=*/(short)0, acc, /*reuse_a=*/false, /*reuse_b=*/false);
    }

    // C/D layout: lane%16 = N col; VGPR v -> M = v + 8*(lane/16)
    const int mb = 8 * (lane >> 4);
    if (colBase < COLS) {
        #pragma unroll
        for (int v = 0; v < 8; ++v)
            proj[(size_t)(rowBase + mb + v) * COLS + ncol] = acc[v];
    } else {
        const int c = ncol - COLS;
        #pragma unroll
        for (int v = 0; v < 8; ++v)
            outAcc[(size_t)(rowBase + mb + v) * COLS + c] = acc[v];
    }
}

// ---------------------------------------------------------------------------
// node_scores: s_src[n,h] = sum_f proj[n,h,f]*a_src[h,f]; same for s_trg
// ---------------------------------------------------------------------------
__global__ void k_node_scores(const float* __restrict__ proj,
                              const float* __restrict__ a_src,
                              const float* __restrict__ a_trg,
                              float* __restrict__ ssrc,
                              float* __restrict__ strg) {
    int i = blockIdx.x * blockDim.x + threadIdx.x;   // over ROWS*HH
    if (i >= ROWS * HH) return;
    int h   = i & (HH - 1);
    int row = i >> 3;
    const float* p = proj + (size_t)row * COLS + h * FOUT;
    const float* as = a_src + h * FOUT;
    const float* at = a_trg + h * FOUT;
    float s0 = 0.0f, s1 = 0.0f;
    #pragma unroll
    for (int f = 0; f < FOUT; ++f) { s0 += p[f] * as[f]; s1 += p[f] * at[f]; }
    ssrc[i] = s0;
    strg[i] = s1;
}

// ordered-uint encoding of float for atomic max over signed floats
__device__ __forceinline__ unsigned f2ord(float f) {
    unsigned b = __float_as_uint(f);
    return (b & 0x80000000u) ? ~b : (b | 0x80000000u);
}
__device__ __forceinline__ float ord2f(unsigned u) {
    unsigned b = (u & 0x80000000u) ? (u & 0x7FFFFFFFu) : ~u;
    return __uint_as_float(b);
}

// ---------------------------------------------------------------------------
// edge_scores: leaky_relu(s_src[src]+s_trg[trg], 0.2); track global max
// ---------------------------------------------------------------------------
__global__ void k_edge_scores(const float* __restrict__ ssrc,
                              const float* __restrict__ strg,
                              const int* __restrict__ eidx,
                              float* __restrict__ scores,
                              unsigned* __restrict__ gmax) {
    __shared__ unsigned smax;
    if (threadIdx.x == 0) smax = 0u;
    __syncthreads();

    int i = blockIdx.x * blockDim.x + threadIdx.x;   // over BT*EDGES*HH
    if (i < TOTAL_EH) {
        int h  = i & (HH - 1);
        int e  = (i >> 3) % EDGES;
        int bt = i / (EDGES * HH);
        int src = eidx[e];
        int trg = eidx[EDGES + e];
        float sc = ssrc[bt * NH + src * HH + h] + strg[bt * NH + trg * HH + h];
        sc = (sc > 0.0f) ? sc : 0.2f * sc;
        scores[i] = sc;
        atomicMax(&smax, f2ord(sc));
    }
    __syncthreads();
    if (threadIdx.x == 0) atomicMax(gmax, smax);
}

// ---------------------------------------------------------------------------
// edge_exp: exp(score - gmax) in place; accumulate softmax denominator
// ---------------------------------------------------------------------------
__global__ void k_edge_exp(float* __restrict__ scores,
                           const int* __restrict__ eidx,
                           float* __restrict__ denom,
                           const unsigned* __restrict__ gmax) {
    int i = blockIdx.x * blockDim.x + threadIdx.x;
    if (i >= TOTAL_EH) return;
    float m = ord2f(*gmax);
    int h  = i & (HH - 1);
    int e  = (i >> 3) % EDGES;
    int bt = i / (EDGES * HH);
    int trg = eidx[EDGES + e];
    float ex = expf(scores[i] - m);
    scores[i] = ex;
    atomicAdd(&denom[bt * NH + trg * HH + h], ex);
}

// ---------------------------------------------------------------------------
// scatter: out[bt,trg,h,f] += proj[bt,src,h,f] * attn(bt,e,h)
// One block per (bt,e); 128 threads cover (h,f).
// ---------------------------------------------------------------------------
__global__ void k_scatter(const float* __restrict__ proj,
                          const float* __restrict__ expv,
                          const float* __restrict__ denom,
                          const int* __restrict__ eidx,
                          float* __restrict__ outAcc) {
    int be = blockIdx.x;                 // 0 .. BT*EDGES-1
    int bt = be / EDGES;
    int e  = be - bt * EDGES;
    int h  = threadIdx.x >> 4;
    int f  = threadIdx.x & 15;
    int src = eidx[e];
    int trg = eidx[EDGES + e];
    float attn = expv[(size_t)be * HH + h] /
                 (denom[bt * NH + trg * HH + h] + 1e-16f);
    float val = proj[((size_t)(bt * NNODE + src)) * COLS + h * FOUT + f] * attn;
    atomicAdd(&outAcc[((size_t)(bt * NNODE + trg)) * COLS + h * FOUT + f], val);
}

// ---------------------------------------------------------------------------
// elu: in-place ELU on output
// ---------------------------------------------------------------------------
__global__ void k_elu(float* __restrict__ out, int n) {
    int i = blockIdx.x * blockDim.x + threadIdx.x;
    if (i >= n) return;
    float v = out[i];
    out[i] = (v > 0.0f) ? v : (expf(v) - 1.0f);
}

// ---------------------------------------------------------------------------
extern "C" void kernel_launch(void* const* d_in, const int* in_sizes, int n_in,
                              void* d_out, int out_size, void* d_ws, size_t ws_size,
                              hipStream_t stream) {
    const float* X    = (const float*)d_in[0];  // [B,T,N,F_in]
    const int*   eidx = (const int*)  d_in[1];  // [2,E]
    const float* Wp   = (const float*)d_in[2];  // [128,64]
    const float* Asrc = (const float*)d_in[3];  // [8,16]
    const float* Atrg = (const float*)d_in[4];  // [8,16]
    const float* Wsk  = (const float*)d_in[5];  // [128,64]
    float* out = (float*)d_out;                 // [ROWS,128]

    // workspace layout (floats)
    float* ws    = (float*)d_ws;
    float* proj  = ws;                                   // ROWS*COLS = 12,288,000
    float* ssrc  = proj  + (size_t)ROWS * COLS;          // 768,000
    float* strg  = ssrc  + (size_t)TOTAL_NHB;            // 768,000
    float* escr  = strg  + (size_t)TOTAL_NHB;            // 6,144,000
    float* denom = escr  + (size_t)TOTAL_EH;             // 768,000
    unsigned* gmax = (unsigned*)(denom + (size_t)TOTAL_NHB);

    // 1) re-init per call (ws/out not re-poisoned between replays)
    k_init<<<(TOTAL_NHB + 255) / 256, 256, 0, stream>>>(denom, gmax);

    // 2) WMMA GEMMs: proj -> ws, skip -> d_out (accumulation base)
    dim3 gg(ROWS / 16, 4);
    k_gemm<<<gg, 128, 0, stream>>>(X, Wp, Wsk, proj, out);

    // 3) per-node attention dots
    k_node_scores<<<(ROWS * HH + 255) / 256, 256, 0, stream>>>(proj, Asrc, Atrg, ssrc, strg);

    // 4) edge scores + global max
    k_edge_scores<<<(TOTAL_EH + 255) / 256, 256, 0, stream>>>(ssrc, strg, eidx, escr, gmax);

    // 5) exp + denominator scatter
    k_edge_exp<<<(TOTAL_EH + 255) / 256, 256, 0, stream>>>(escr, eidx, denom, gmax);

    // 6) weighted feature scatter into out (on top of skip)
    k_scatter<<<BT * EDGES, 128, 0, stream>>>(proj, escr, denom, eidx, out);

    // 7) ELU in place
    k_elu<<<(TOTAL_OUT + 255) / 256, 256, 0, stream>>>(out, TOTAL_OUT);
}
